// BidirectionalMambaContextBlock_15934328668529
// MI455X (gfx1250) — matrix operand order repaired and checked by reading, not compiled, not run on hardware
//
#include <hip/hip_runtime.h>
#include <math.h>

// ---------------------------------------------------------------------------
// Types for CDNA5 WMMA (gfx1250, wave32)
// ---------------------------------------------------------------------------
typedef __bf16 bf16_t;
typedef bf16_t bf16x8  __attribute__((ext_vector_type(8)));
typedef bf16_t bf16x16 __attribute__((ext_vector_type(16)));
typedef float  f32x8   __attribute__((ext_vector_type(8)));
typedef int    i32x4v  __attribute__((vector_size(16)));   // matches builtin's V4i

#define AS1 __attribute__((address_space(1)))
#define AS3 __attribute__((address_space(3)))

#if defined(__HIP_DEVICE_COMPILE__) && __has_builtin(__builtin_amdgcn_global_load_async_to_lds_b128)
#define USE_ASYNC_LDS 1
#else
#define USE_ASYNC_LDS 0
#endif

#if USE_ASYNC_LDS
__device__ __forceinline__ void async_copy_b128(const bf16_t* g, bf16_t* l) {
    // param1: AS1 (global) ptr to int4 ; param2: AS3 (LDS) ptr to int4 ;
    // param3: imm offset ; param4: imm cpol
    __builtin_amdgcn_global_load_async_to_lds_b128(
        (AS1 i32x4v*)(AS1 bf16_t*)g,
        (AS3 i32x4v*)(AS3 bf16_t*)l, 0, 0);
}
#if __has_builtin(__builtin_amdgcn_s_wait_asynccnt)
#define WAIT_ASYNC(n) __builtin_amdgcn_s_wait_asynccnt((unsigned short)(n))
#else
#define WAIT_ASYNC(n)                                                    \
    do {                                                                 \
        if ((n) == 0) asm volatile("s_wait_asynccnt 0x0" ::: "memory");  \
        else          asm volatile("s_wait_asynccnt 0x2" ::: "memory");  \
    } while (0)
#endif
#endif

#define LSEQ   1024
#define BATCH  8
#define BL     (BATCH * LSEQ)
#define DMODEL 512
#define DINNER 1024
#define NSTATE 16
#define NPROJ  64
#define DTRANK 32

// Padded LDS row stride (elements) for the staged B tile: 80B/row keeps the 16
// rows of one fragment in distinct bank groups for the 32B ds reads.
#define BT_STRIDE 40

__device__ __forceinline__ float gelu_f(float x)     { return 0.5f * x * (1.0f + erff(x * 0.70710678118f)); }
__device__ __forceinline__ float silu_f(float x)     { return x / (1.0f + __expf(-x)); }
__device__ __forceinline__ float softplus_f(float x) { return (x > 20.0f) ? x : log1pf(__expf(x)); }

// ---------------------------------------------------------------------------
// f32 -> bf16 conversion (weights, one pass)
// ---------------------------------------------------------------------------
__global__ void k_cvt_bf16(const float* __restrict__ s, bf16_t* __restrict__ d, int n) {
    int i = blockIdx.x * blockDim.x + threadIdx.x;
    if (i < n) d[i] = (bf16_t)s[i];
}

// ---------------------------------------------------------------------------
// Generic WMMA GEMM:  out[M,N] = act(A[M,K](bf16) * W[N,K]^T(bf16) + bias) + addsrc
// Block = 128 threads (4 waves). Each wave: 16(M) x 64(N) tile, K step = 32.
// The 64x32 B tile is shared by all 4 waves and staged through a
// double-buffered LDS tile via the gfx1250 async global->LDS path
// (ASYNCcnt + s_wait_asynccnt); fallback streams it from global.
// Fragment layouts used here (wave32):
//   A 16x32 bf16 : lane m = lane&15; half = lane>>4
//                  elems 0..15 <- K = 16*half + 0..15          (contiguous 32B)
//   B 32x16 bf16 : lane n = lane&15; khalf = lane>>4
//                  elems 0..15 <- K = 16*khalf + 0..15          (contiguous 32B)
//   C/D 16x16 f32: lane n = lane&15; reg r -> m = r + 8*(lane>>4)
// revL != 0 reverses A's row index within each length-revL segment.
// ---------------------------------------------------------------------------
__global__ __launch_bounds__(128)
void k_gemm(const bf16_t* __restrict__ A, const bf16_t* __restrict__ Wt,
            int M, int N, int K, int revL,
            const float* __restrict__ bias, int act,
            const float* __restrict__ addsrc,
            float* __restrict__ outf, bf16_t* __restrict__ outbf)
{
    const int wave = threadIdx.x >> 5;
    const int lane = threadIdx.x & 31;
    const int row  = lane & 15;
    const int half = lane >> 4;
    const int tileN = blockIdx.x * 64;
    const int tileM = blockIdx.y * 64 + wave * 16;

    int grow = tileM + row;
    if (revL) { int l = grow % revL; grow += (revL - 1 - l) - l; }
    const bf16_t* arow = A + (size_t)grow * K;

    f32x8 acc0 = {0.f,0.f,0.f,0.f,0.f,0.f,0.f,0.f};
    f32x8 acc1 = acc0, acc2 = acc0, acc3 = acc0;

#if USE_ASYNC_LDS
    // ---- async double-buffered B tile in LDS --------------------------------
    __shared__ __align__(16) bf16_t btile[2][64 * BT_STRIDE];

    const int S = K >> 5;
    const int idx0 = (int)threadIdx.x * 2;      // two b128 chunks per thread
    // chunk idx -> row r = idx>>2, 16B sub-chunk c = idx&3
    const int r0 = idx0 >> 2,       c0 = idx0 & 3;
    const int r1 = (idx0 + 1) >> 2, c1 = (idx0 + 1) & 3;

    {   // prologue: stage chunk 0 into buffer 0
        async_copy_b128(Wt + (size_t)(tileN + r0) * K + c0 * 8,
                        &btile[0][r0 * BT_STRIDE + c0 * 8]);
        async_copy_b128(Wt + (size_t)(tileN + r1) * K + c1 * 8,
                        &btile[0][r1 * BT_STRIDE + c1 * 8]);
    }

    for (int s = 0; s < S; ++s) {
        const int kk = s << 5;
        if (s + 1 < S) {
            const int kn = kk + 32;
            async_copy_b128(Wt + (size_t)(tileN + r0) * K + kn + c0 * 8,
                            &btile[(s + 1) & 1][r0 * BT_STRIDE + c0 * 8]);
            async_copy_b128(Wt + (size_t)(tileN + r1) * K + kn + c1 * 8,
                            &btile[(s + 1) & 1][r1 * BT_STRIDE + c1 * 8]);
            WAIT_ASYNC(2);          // in-order completion: chunk s is resident
        } else {
            WAIT_ASYNC(0);
        }
        __syncthreads();            // tile visible to all 4 waves

        __builtin_prefetch(arow + kk + 8 * half + 128, 0, 1);
        bf16x16 afrag = *(const bf16x16*)(arow + kk + 16 * half);

        const bf16_t* lb = &btile[s & 1][0];
        bf16x16 bf0 = *(const bf16x16*)(lb + ( 0 + row) * BT_STRIDE + half * 16);
        bf16x16 bf1 = *(const bf16x16*)(lb + (16 + row) * BT_STRIDE + half * 16);
        bf16x16 bf2 = *(const bf16x16*)(lb + (32 + row) * BT_STRIDE + half * 16);
        bf16x16 bf3 = *(const bf16x16*)(lb + (48 + row) * BT_STRIDE + half * 16);

        acc0 = __builtin_amdgcn_wmma_f32_16x16x32_bf16(false, afrag, false, bf0, (short)0, acc0, false, false);
        acc1 = __builtin_amdgcn_wmma_f32_16x16x32_bf16(false, afrag, false, bf1, (short)0, acc1, false, false);
        acc2 = __builtin_amdgcn_wmma_f32_16x16x32_bf16(false, afrag, false, bf2, (short)0, acc2, false, false);
        acc3 = __builtin_amdgcn_wmma_f32_16x16x32_bf16(false, afrag, false, bf3, (short)0, acc3, false, false);

        __syncthreads();            // done reading buf[s&1] before it is reused
    }
#else
    // ---- fallback: stream B straight from global ----------------------------
    const bf16_t* b0 = Wt + (size_t)(tileN +  0 + row) * K + 16 * half;
    const bf16_t* b1 = Wt + (size_t)(tileN + 16 + row) * K + 16 * half;
    const bf16_t* b2 = Wt + (size_t)(tileN + 32 + row) * K + 16 * half;
    const bf16_t* b3 = Wt + (size_t)(tileN + 48 + row) * K + 16 * half;

    for (int k = 0; k < K; k += 32) {
        __builtin_prefetch(arow + k + 8 * half + 128, 0, 1);
        __builtin_prefetch(b0 + k + 128, 0, 1);

        bf16x16 afrag = *(const bf16x16*)(arow + k + 16 * half);

        bf16x16 bf0 = *(const bf16x16*)(b0 + k);
        bf16x16 bf1 = *(const bf16x16*)(b1 + k);
        bf16x16 bf2 = *(const bf16x16*)(b2 + k);
        bf16x16 bf3 = *(const bf16x16*)(b3 + k);

        acc0 = __builtin_amdgcn_wmma_f32_16x16x32_bf16(false, afrag, false, bf0, (short)0, acc0, false, false);
        acc1 = __builtin_amdgcn_wmma_f32_16x16x32_bf16(false, afrag, false, bf1, (short)0, acc1, false, false);
        acc2 = __builtin_amdgcn_wmma_f32_16x16x32_bf16(false, afrag, false, bf2, (short)0, acc2, false, false);
        acc3 = __builtin_amdgcn_wmma_f32_16x16x32_bf16(false, afrag, false, bf3, (short)0, acc3, false, false);
    }
#endif

    f32x8 accs[4] = {acc0, acc1, acc2, acc3};
    const int mbase = tileM + 8 * half;
#pragma unroll
    for (int t = 0; t < 4; ++t) {
        const int n = tileN + t * 16 + row;
        const float bb = bias ? bias[n] : 0.0f;
#pragma unroll
        for (int r = 0; r < 8; ++r) {
            const int m = mbase + r;
            float v = accs[t][r] + bb;
            if      (act == 1) v = gelu_f(v);
            else if (act == 2) v = silu_f(v);
            else if (act == 3) v = softplus_f(v);
            const size_t o = (size_t)m * N + n;
            if (addsrc) v += addsrc[o];
            if (outf)  outf[o]  = v;
            if (outbf) outbf[o] = (bf16_t)v;
        }
    }
}

// ---------------------------------------------------------------------------
// Positional MLP stage 1: hid[l,c] = bf16(gelu(pos6(l) . w1[c,:] + b1[c]))
// ---------------------------------------------------------------------------
__global__ __launch_bounds__(512)
void k_posmlp1(const float* __restrict__ w1, const float* __restrict__ b1,
               bf16_t* __restrict__ hid)
{
    const int l = blockIdx.x;
    const int c = threadIdx.x;
    const int yi = l >> 5, xi = l & 31;
    const float yv = ((yi + 0.5f) / 32.0f) * 2.0f - 1.0f;
    const float xv = ((xi + 0.5f) / 32.0f) * 2.0f - 1.0f;
    const float PI = 3.14159265358979323846f;
    float p[6] = { yv, xv, sinf(PI * yv), cosf(PI * yv), sinf(PI * xv), cosf(PI * xv) };
    float acc = b1[c];
#pragma unroll
    for (int j = 0; j < 6; ++j) acc += p[j] * w1[c * 6 + j];
    hid[(size_t)l * DMODEL + c] = (bf16_t)gelu_f(acc);
}

// ---------------------------------------------------------------------------
// scan_in = bf16( LN(tokens)*g + b + pos[l] ); one block per token, 256 thr.
// ---------------------------------------------------------------------------
__global__ __launch_bounds__(256)
void k_ln_add_pos(const float* __restrict__ tok, const float* __restrict__ g,
                  const float* __restrict__ bt, const float* __restrict__ pos,
                  bf16_t* __restrict__ out)
{
    __shared__ float s1[256], s2[256];
    const int tid = threadIdx.x;
    const int t = blockIdx.x;
    const int l = t & (LSEQ - 1);
    const float* xr = tok + (size_t)t * DMODEL;
    const float x0 = xr[tid], x1 = xr[tid + 256];
    s1[tid] = x0 + x1;
    s2[tid] = x0 * x0 + x1 * x1;
    __syncthreads();
    for (int st = 128; st > 0; st >>= 1) {
        if (tid < st) { s1[tid] += s1[tid + st]; s2[tid] += s2[tid + st]; }
        __syncthreads();
    }
    const float mean = s1[0] * (1.0f / DMODEL);
    const float var  = s2[0] * (1.0f / DMODEL) - mean * mean;
    const float inv  = rsqrtf(var + 1e-5f);
    const size_t o = (size_t)t * DMODEL;
    out[o + tid]       = (bf16_t)((x0 - mean) * inv * g[tid]       + bt[tid]       + pos[(size_t)l * DMODEL + tid]);
    out[o + tid + 256] = (bf16_t)((x1 - mean) * inv * g[tid + 256] + bt[tid + 256] + pos[(size_t)l * DMODEL + tid + 256]);
}

// ---------------------------------------------------------------------------
// Causal depthwise conv (DCONV=4) + SiLU over xin = xz[..., :1024]
// ---------------------------------------------------------------------------
__global__ __launch_bounds__(256)
void k_conv(const float* __restrict__ xz, const float* __restrict__ cw,
            const float* __restrict__ cb, float* __restrict__ xc,
            bf16_t* __restrict__ xcbf)
{
    const int t = blockIdx.x;          // token index in [0, B*L)
    const int l = t & (LSEQ - 1);
    for (int c = threadIdx.x; c < DINNER; c += 256) {
        float acc = cb[c];
#pragma unroll
        for (int k = 0; k < 4; ++k) {
            const int ls = l - 3 + k;
            if (ls >= 0) acc += xz[(size_t)(t - 3 + k) * 2048 + c] * cw[c * 4 + k];
        }
        const float v = silu_f(acc);
        xc[(size_t)t * DINNER + c]   = v;
        xcbf[(size_t)t * DINNER + c] = (bf16_t)v;
    }
}

// ---------------------------------------------------------------------------
// Extract dt (bf16) from proj[..., :32]
// ---------------------------------------------------------------------------
__global__ void k_dtcast(const float* __restrict__ proj, bf16_t* __restrict__ dt) {
    const int i = blockIdx.x * blockDim.x + threadIdx.x;
    if (i < BL * DTRANK) dt[i] = (bf16_t)proj[(size_t)(i >> 5) * NPROJ + (i & 31)];
}

// ---------------------------------------------------------------------------
// Selective scan. One thread per (b, di) channel; h[16] in registers.
// Fused epilogue: y_bf = bf16( (ys + xc*D) * silu(z) )
// Grid = BATCH*4 blocks x 256 threads.
// ---------------------------------------------------------------------------
__global__ __launch_bounds__(256)
void k_scan(const float* __restrict__ delta, const float* __restrict__ xc,
            const float* __restrict__ xz, const float* __restrict__ proj,
            const float* __restrict__ A_log, const float* __restrict__ Dp,
            bf16_t* __restrict__ ybf)
{
    const int b  = blockIdx.x >> 2;
    const int di = (blockIdx.x & 3) * 256 + threadIdx.x;
    float a[NSTATE], h[NSTATE];
#pragma unroll
    for (int s = 0; s < NSTATE; ++s) {
        a[s] = -expf(A_log[(size_t)di * NSTATE + s]);
        h[s] = 0.0f;
    }
    const float Dv = Dp[di];
    __shared__ float bc[32];
    const size_t baseBL = (size_t)b * LSEQ;
    for (int l = 0; l < LSEQ; ++l) {
        const size_t t = baseBL + l;
        if (threadIdx.x < 32) bc[threadIdx.x] = proj[t * NPROJ + DTRANK + threadIdx.x];
        __syncthreads();
        const float dlt = delta[t * DINNER + di];
        const float x   = xc[t * DINNER + di];
        const float z   = xz[t * 2048 + DINNER + di];
        const float dx  = dlt * x;
        float y = 0.0f;
#pragma unroll
        for (int s = 0; s < NSTATE; ++s) {
            h[s] = h[s] * __expf(dlt * a[s]) + dx * bc[s];
            y += h[s] * bc[NSTATE + s];
        }
        ybf[t * DINNER + di] = (bf16_t)((y + x * Dv) * silu_f(z));
        __syncthreads();
    }
}

// ---------------------------------------------------------------------------
// cat = bf16([fwd, reverse_l(bwd), pos])
// ---------------------------------------------------------------------------
__global__ __launch_bounds__(512)
void k_cat(const float* __restrict__ fwd, const float* __restrict__ bwd,
           const float* __restrict__ pos, bf16_t* __restrict__ cat)
{
    const int t = blockIdx.x;
    const int l = t & (LSEQ - 1);
    const int b = t >> 10;
    const int c = threadIdx.x;
    const size_t o = (size_t)t * 1536;
    cat[o + c]        = (bf16_t)fwd[(size_t)t * DMODEL + c];
    cat[o + 512 + c]  = (bf16_t)bwd[(((size_t)b << 10) + (LSEQ - 1 - l)) * DMODEL + c];
    cat[o + 1024 + c] = (bf16_t)pos[(size_t)l * DMODEL + c];
}

// ---------------------------------------------------------------------------
// Two chained LayerNorms: d = LN(x)*g1+b1 (kept f32, residual source);
//                         f = bf16( LN(d)*g2+b2 )   (FFN input)
// ---------------------------------------------------------------------------
__global__ __launch_bounds__(256)
void k_double_ln(const float* __restrict__ x, const float* __restrict__ g1,
                 const float* __restrict__ bb1, const float* __restrict__ g2,
                 const float* __restrict__ bb2, float* __restrict__ dout,
                 bf16_t* __restrict__ fbf)
{
    __shared__ float s1[256], s2[256];
    const int tid = threadIdx.x;
    const float* xr = x + (size_t)blockIdx.x * DMODEL;
    const float x0 = xr[tid], x1 = xr[tid + 256];
    s1[tid] = x0 + x1;
    s2[tid] = x0 * x0 + x1 * x1;
    __syncthreads();
    for (int st = 128; st > 0; st >>= 1) {
        if (tid < st) { s1[tid] += s1[tid + st]; s2[tid] += s2[tid + st]; }
        __syncthreads();
    }
    float mean = s1[0] * (1.0f / DMODEL);
    float var  = s2[0] * (1.0f / DMODEL) - mean * mean;
    float inv  = rsqrtf(var + 1e-5f);
    const float d0 = (x0 - mean) * inv * g1[tid]       + bb1[tid];
    const float d1 = (x1 - mean) * inv * g1[tid + 256] + bb1[tid + 256];
    const size_t o = (size_t)blockIdx.x * DMODEL;
    dout[o + tid]       = d0;
    dout[o + tid + 256] = d1;
    __syncthreads();
    s1[tid] = d0 + d1;
    s2[tid] = d0 * d0 + d1 * d1;
    __syncthreads();
    for (int st = 128; st > 0; st >>= 1) {
        if (tid < st) { s1[tid] += s1[tid + st]; s2[tid] += s2[tid + st]; }
        __syncthreads();
    }
    mean = s1[0] * (1.0f / DMODEL);
    var  = s2[0] * (1.0f / DMODEL) - mean * mean;
    inv  = rsqrtf(var + 1e-5f);
    fbf[o + tid]       = (bf16_t)((d0 - mean) * inv * g2[tid]       + bb2[tid]);
    fbf[o + tid + 256] = (bf16_t)((d1 - mean) * inv * g2[tid + 256] + bb2[tid + 256]);
}

// ---------------------------------------------------------------------------
// Host side
// ---------------------------------------------------------------------------
extern "C" void kernel_launch(void* const* d_in, const int* in_sizes, int n_in,
                              void* d_out, int out_size, void* d_ws, size_t ws_size,
                              hipStream_t stream)
{
    (void)in_sizes; (void)n_in; (void)out_size; (void)ws_size;

    const float* tokens   = (const float*)d_in[0];
    const float* in_g     = (const float*)d_in[3];
    const float* in_b     = (const float*)d_in[4];
    const float* pos_w1   = (const float*)d_in[5];
    const float* pos_b1   = (const float*)d_in[6];
    const float* pos_w2   = (const float*)d_in[7];
    const float* pos_b2   = (const float*)d_in[8];
    const float* m_in_w   = (const float*)d_in[9];
    const float* m_conv_w = (const float*)d_in[10];
    const float* m_conv_b = (const float*)d_in[11];
    const float* m_xproj  = (const float*)d_in[12];
    const float* m_dt_w   = (const float*)d_in[13];
    const float* m_dt_b   = (const float*)d_in[14];
    const float* m_A_log  = (const float*)d_in[15];
    const float* m_D      = (const float*)d_in[16];
    const float* m_out_w  = (const float*)d_in[17];
    const float* mix_w    = (const float*)d_in[18];
    const float* mix_b    = (const float*)d_in[19];
    const float* dn_g     = (const float*)d_in[20];
    const float* dn_b     = (const float*)d_in[21];
    const float* fn_g     = (const float*)d_in[22];
    const float* fn_b     = (const float*)d_in[23];
    const float* ffn_w1   = (const float*)d_in[24];
    const float* ffn_b1   = (const float*)d_in[25];
    const float* ffn_w2   = (const float*)d_in[26];
    const float* ffn_b2   = (const float*)d_in[27];

    char* base = (char*)d_ws;
    size_t off = 0;
    auto take = [&](size_t bytes) -> char* {
        char* p = base + off;
        off = (off + bytes + 255) & ~(size_t)255;
        return p;
    };

    // Persistent bf16 weights + small buffers
    bf16_t* w_in_bf  = (bf16_t*)take((size_t)2 * 2048 * 512 * 2);
    bf16_t* w_xp_bf  = (bf16_t*)take((size_t)2 * 64 * 1024 * 2);
    bf16_t* w_dt_bf  = (bf16_t*)take((size_t)2 * 1024 * 32 * 2);
    bf16_t* w_out_bf = (bf16_t*)take((size_t)2 * 512 * 1024 * 2);
    bf16_t* w_mix_bf = (bf16_t*)take((size_t)512 * 1536 * 2);
    bf16_t* w_f1_bf  = (bf16_t*)take((size_t)1024 * 512 * 2);
    bf16_t* w_f2_bf  = (bf16_t*)take((size_t)512 * 1024 * 2);
    bf16_t* w_p2_bf  = (bf16_t*)take((size_t)512 * 512 * 2);
    bf16_t* pos_hid  = (bf16_t*)take((size_t)LSEQ * DMODEL * 2);
    float*  pos_f    = (float*) take((size_t)LSEQ * DMODEL * 4);
    bf16_t* scan_bf  = (bf16_t*)take((size_t)BL * DMODEL * 2);
    float*  dir_out  = (float*) take((size_t)2 * BL * DMODEL * 4);

    // Reused scratch arena
    const size_t SZ_XZ   = (size_t)BL * 2048 * 4;
    const size_t SZ_XC   = (size_t)BL * DINNER * 4;
    const size_t SZ_XCBF = (size_t)BL * DINNER * 2;
    const size_t SZ_PROJ = (size_t)BL * NPROJ * 4;
    const size_t SZ_DTBF = (size_t)BL * DTRANK * 2;
    const size_t SZ_DEL  = (size_t)BL * DINNER * 4;
    const size_t SZ_YBF  = (size_t)BL * DINNER * 2;
    char* scratch = take(SZ_XZ + SZ_XC + SZ_XCBF + SZ_PROJ + SZ_DTBF + SZ_DEL + SZ_YBF);

    float*  xz    = (float*)  scratch;
    float*  xc    = (float*) (scratch + SZ_XZ);
    bf16_t* xc_bf = (bf16_t*)(scratch + SZ_XZ + SZ_XC);
    float*  proj  = (float*) (scratch + SZ_XZ + SZ_XC + SZ_XCBF);
    bf16_t* dt_bf = (bf16_t*)(scratch + SZ_XZ + SZ_XC + SZ_XCBF + SZ_PROJ);
    float*  delta = (float*) (scratch + SZ_XZ + SZ_XC + SZ_XCBF + SZ_PROJ + SZ_DTBF);
    bf16_t* y_bf  = (bf16_t*)(scratch + SZ_XZ + SZ_XC + SZ_XCBF + SZ_PROJ + SZ_DTBF + SZ_DEL);

    // Tail-stage aliases (mamba scratch is dead by then)
    bf16_t* cat_bf   = (bf16_t*) scratch;                                              // 24 MB
    float*  mix_f    = (float*) (scratch + (size_t)BL * 1536 * 2);                     // 16 MB
    bf16_t* fhid_bf  = (bf16_t*)(scratch + (size_t)BL * 1536 * 2 + (size_t)BL * 512 * 4);
    float*  delta_ln = (float*) (scratch + SZ_XZ);                                     // xc region
    bf16_t* fnorm_bf = (bf16_t*)(scratch + SZ_XZ + (size_t)BL * 512 * 4);

    auto cvt = [&](const float* s, bf16_t* d, int n) {
        k_cvt_bf16<<<(n + 255) / 256, 256, 0, stream>>>(s, d, n);
    };
    auto gemm = [&](const bf16_t* A, const bf16_t* Wt, int M, int N, int K, int revL,
                    const float* bias, int act, const float* addsrc,
                    float* outf, bf16_t* outbf) {
        dim3 g(N / 64, M / 64), blk(128);
        k_gemm<<<g, blk, 0, stream>>>(A, Wt, M, N, K, revL, bias, act, addsrc, outf, outbf);
    };

    // 1) Weight conversions
    cvt(m_in_w,  w_in_bf,  2 * 2048 * 512);
    cvt(m_xproj, w_xp_bf,  2 * 64 * 1024);
    cvt(m_dt_w,  w_dt_bf,  2 * 1024 * 32);
    cvt(m_out_w, w_out_bf, 2 * 512 * 1024);
    cvt(mix_w,   w_mix_bf, 512 * 1536);
    cvt(ffn_w1,  w_f1_bf,  1024 * 512);
    cvt(ffn_w2,  w_f2_bf,  512 * 1024);
    cvt(pos_w2,  w_p2_bf,  512 * 512);

    // 2) Positional MLP: pos = gelu(pos6@w1^T+b1)@w2^T+b2
    k_posmlp1<<<LSEQ, 512, 0, stream>>>(pos_w1, pos_b1, pos_hid);
    gemm(pos_hid, w_p2_bf, LSEQ, DMODEL, DMODEL, 0, pos_b2, 0, nullptr, pos_f, nullptr);

    // 3) scan_in = bf16(LN(tokens) + pos)
    k_ln_add_pos<<<BL, 256, 0, stream>>>(tokens, in_g, in_b, pos_f, scan_bf);

    // 4) Both mamba directions (backward folded via revL on the in-projection)
    for (int d = 0; d < 2; ++d) {
        gemm(scan_bf, w_in_bf + (size_t)d * 2048 * 512, BL, 2048, 512,
             d ? LSEQ : 0, nullptr, 0, nullptr, xz, nullptr);
        k_conv<<<BL, 256, 0, stream>>>(xz, m_conv_w + (size_t)d * DINNER * 4,
                                       m_conv_b + (size_t)d * DINNER, xc, xc_bf);
        gemm(xc_bf, w_xp_bf + (size_t)d * 64 * 1024, BL, NPROJ, DINNER,
             0, nullptr, 0, nullptr, proj, nullptr);
        k_dtcast<<<(BL * DTRANK + 255) / 256, 256, 0, stream>>>(proj, dt_bf);
        gemm(dt_bf, w_dt_bf + (size_t)d * 1024 * 32, BL, DINNER, DTRANK,
             0, m_dt_b + (size_t)d * DINNER, 3, nullptr, delta, nullptr);
        k_scan<<<BATCH * 4, 256, 0, stream>>>(delta, xc, xz, proj,
                                              m_A_log + (size_t)d * DINNER * NSTATE,
                                              m_D + (size_t)d * DINNER, y_bf);
        gemm(y_bf, w_out_bf + (size_t)d * 512 * 1024, BL, DMODEL, DINNER,
             0, nullptr, 0, nullptr, dir_out + (size_t)d * BL * DMODEL, nullptr);
    }

    // 5) mix + double LN + FFN (+residual into d_out)
    k_cat<<<BL, 512, 0, stream>>>(dir_out, dir_out + (size_t)BL * DMODEL, pos_f, cat_bf);
    gemm(cat_bf, w_mix_bf, BL, DMODEL, 1536, 0, mix_b, 0, nullptr, mix_f, nullptr);
    k_double_ln<<<BL, 256, 0, stream>>>(mix_f, dn_g, dn_b, fn_g, fn_b, delta_ln, fnorm_bf);
    gemm(fnorm_bf, w_f1_bf, BL, DINNER, DMODEL, 0, ffn_b1, 1, nullptr, nullptr, fhid_bf);
    gemm(fhid_bf, w_f2_bf, BL, DMODEL, DINNER, 0, ffn_b2, 0, delta_ln, (float*)d_out, nullptr);
}
